// Model_75728863363281
// MI455X (gfx1250) — matrix operand order repaired and checked by reading, not compiled, not run on hardware
//
#include <hip/hip_runtime.h>

typedef __attribute__((ext_vector_type(16))) _Float16 v16h;
typedef __attribute__((ext_vector_type(8)))  float    v8f;

#define N_NODES 20000
#define N_EDGES 320000
#define HID     32
#define PERIODS 12
#define OUTP    12
#define FEATS   24   // F_IN * PERIODS, contiguous per node in X[N,2,12]
#define NCONST  204  // 64 Cz + 32 cz + 64 Ch + 32 ch + 12 probs

// ---------------- degree / normalization ----------------

__global__ void k_init_deg(float* __restrict__ deg) {
  int n = blockIdx.x * blockDim.x + threadIdx.x;
  if (n < N_NODES) deg[n] = 1.0f;  // self loop weight
}

__global__ void k_deg_edges(const long long* __restrict__ ei, float* __restrict__ deg) {
  int e = blockIdx.x * blockDim.x + threadIdx.x;
  if (e < N_EDGES) {
    int d = (int)ei[N_EDGES + e];  // dst row
    atomicAdd(&deg[d], 1.0f);
  }
}

// dinv[n] = rsqrt(deg); AX[n,:] = dinv^2 * X[n,:]  (self-loop contribution)
__global__ void k_dinv_selfinit(const float* __restrict__ X,
                                float* __restrict__ deg_dinv,
                                float* __restrict__ AX) {
  int n = blockIdx.x * blockDim.x + threadIdx.x;
  if (n >= N_NODES) return;
  float dg = deg_dinv[n];
  float di = (dg > 0.0f) ? rsqrtf(dg) : 0.0f;
  deg_dinv[n] = di;
  float w = di * di;
  const float4* xr = (const float4*)(X + (size_t)n * FEATS);
  float4* ar = (float4*)(AX + (size_t)n * FEATS);
#pragma unroll
  for (int q = 0; q < 6; ++q) {
    float4 v = xr[q];
    v.x *= w; v.y *= w; v.z *= w; v.w *= w;
    ar[q] = v;
  }
}

// AX[dst,:] += dinv[src]*dinv[dst] * X[src,:]
__global__ void k_scatter(const long long* __restrict__ ei,
                          const float* __restrict__ X,
                          const float* __restrict__ dinv,
                          float* __restrict__ AX) {
  int e = blockIdx.x * blockDim.x + threadIdx.x;
  if (e >= N_EDGES) return;
  int s = (int)ei[e];
  int d = (int)ei[N_EDGES + e];
  float nrm = dinv[s] * dinv[d];
  const float4* xr = (const float4*)(X + (size_t)s * FEATS);
  float* ar = AX + (size_t)d * FEATS;
#pragma unroll
  for (int q = 0; q < 6; ++q) {
    float4 v = xr[q];
    atomicAdd(ar + 4 * q + 0, nrm * v.x);
    atomicAdd(ar + 4 * q + 1, nrm * v.y);
    atomicAdd(ar + 4 * q + 2, nrm * v.z);
    atomicAdd(ar + 4 * q + 3, nrm * v.w);
  }
}

// ---------------- fused constants ----------------
// C layout (floats): [0,64) Cz[f*32+j]  [64,96) cz  [96,160) Ch  [160,192) ch  [192,204) probs
__global__ void k_consts(const float* __restrict__ czW, const float* __restrict__ czb,
                         const float* __restrict__ chW, const float* __restrict__ chb,
                         const float* __restrict__ lzW, const float* __restrict__ lzb,
                         const float* __restrict__ lhW, const float* __restrict__ lhb,
                         const float* __restrict__ att, float* __restrict__ C) {
  int t = threadIdx.x;
  if (t < 64) {                       // Cz = conv_z_W @ lin_z_W[:32], Ch likewise
    int f = t >> 5, j = t & 31;
    float sz = 0.0f, sh = 0.0f;
    for (int i = 0; i < HID; ++i) {
      sz += czW[f * HID + i] * lzW[i * HID + j];
      sh += chW[f * HID + i] * lhW[i * HID + j];
    }
    C[f * HID + j]      = sz;
    C[96 + f * HID + j] = sh;
  }
  if (t < 32) {                       // fused biases
    int j = t;
    float sz = lzb[j], sh = lhb[j];
    for (int i = 0; i < HID; ++i) {
      sz += czb[i] * lzW[i * HID + j];
      sh += chb[i] * lhW[i * HID + j];
    }
    C[64 + j]  = sz;
    C[160 + j] = sh;
  }
  if (t == 0) {                       // softmax(att)
    float m = att[0];
    for (int p = 1; p < PERIODS; ++p) m = fmaxf(m, att[p]);
    float ex[PERIODS]; float s = 0.0f;
    for (int p = 0; p < PERIODS; ++p) { ex[p] = __expf(att[p] - m); s += ex[p]; }
    for (int p = 0; p < PERIODS; ++p) C[192 + p] = ex[p] / s;
  }
}

// ---------------- dense epilogue: gates + attention accum + WMMA projection ----------------
// One wave32 per 16-node tile. Accumulators computed directly in WMMA A-fragment order.
// Constants staged once per block in LDS; all bounds handling is branch-free (clamped
// index + 0/1 mask) so loads clause up and EXEC stays all-1s at the WMMA.
__global__ void __launch_bounds__(256) k_dense(const float* __restrict__ AX,
                                               const float* __restrict__ C,
                                               const float* __restrict__ outW,
                                               const float* __restrict__ outB,
                                               float* __restrict__ out) {
  __shared__ float sC[NCONST];
  if (threadIdx.x < NCONST) sC[threadIdx.x] = C[threadIdx.x];
  __syncthreads();

  const int lane = threadIdx.x & 31;
  const int wave = threadIdx.x >> 5;
  const int tile = blockIdx.x * (blockDim.x >> 5) + wave;
  const int n_tiles = N_NODES / 16;

  if (tile < n_tiles) {                        // wave-uniform: EXEC all-1s inside
    const int half = lane >> 4;                // 0: K/j-slots {0..7,16..23}; 1: {8..15,24..31}
    const int m    = lane & 15;                // node-in-tile (A rows) == output column (C/D)
    const int node = tile * 16 + m;

    // Per-lane gate constants in A-fragment slot order: j(s) = (s>>3)*16 + half*8 + (s&7)
    float Cz0[16], Cz1[16], czv[16], Ch0[16], Ch1[16], chv[16];
#pragma unroll
    for (int s = 0; s < 16; ++s) {
      int j = ((s >> 3) << 4) + (half << 3) + (s & 7);
      Cz0[s] = sC[j];       Cz1[s] = sC[32 + j];  czv[s] = sC[64 + j];
      Ch0[s] = sC[96 + j];  Ch1[s] = sC[128 + j]; chv[s] = sC[160 + j];
    }

    // Aggregated features for this node: a0[p]=AX[n,0,p], a1[p]=AX[n,1,p]
    const float* xr = AX + (size_t)node * FEATS;
    float a0[PERIODS], a1[PERIODS];
#pragma unroll
    for (int p = 0; p < PERIODS; ++p) { a0[p] = xr[p]; a1[p] = xr[PERIODS + p]; }

    float acc[16];
#pragma unroll
    for (int s = 0; s < 16; ++s) acc[s] = 0.0f;

    for (int p = 0; p < PERIODS; ++p) {
      float pr = sC[192 + p];
      float x0 = a0[p], x1 = a1[p];
#pragma unroll
      for (int s = 0; s < 16; ++s) {
        float u  = fmaf(x0, Cz0[s], fmaf(x1, Cz1[s], czv[s]));   // gate z pre-act
        float v  = fmaf(x0, Ch0[s], fmaf(x1, Ch1[s], chv[s]));   // candidate pre-act
        float e2 = __expf(2.0f * v);
        float th = (e2 - 1.0f) / (e2 + 1.0f);                    // tanh(v)
        float h  = th / (1.0f + __expf(u));                      // (1-sigmoid(u))*tanh(v)
        acc[s]   = fmaf(pr, h, acc[s]);
      }
    }

    // A fragment: relu(acc) as f16, already in fragment order
    v16h a;
#pragma unroll
    for (int s = 0; s < 16; ++s) a[s] = (_Float16)fmaxf(acc[s], 0.0f);

    // B fragment: out_W [32 x 12] zero-padded to 32 x 16 — branch-free:
    // clamp column index, scale by 0/1 keep mask -> one clause of loads.
    const int   kbase = half ? 16 : 0;
    const int   mc    = (m < OUTP) ? m : (OUTP - 1);
    const float keep  = (m < OUTP) ? 1.0f : 0.0f;
    v16h b;
#pragma unroll
    for (int s = 0; s < 16; ++s)
      b[s] = (_Float16)(outW[(((s >> 3) << 4) + (half << 3) + (s & 7)) * OUTP + mc] * keep);

    // C fragment: bias broadcast down column n=m
    float bias = outB[mc] * keep;
    v8f c;
#pragma unroll
    for (int r = 0; r < 8; ++r) c[r] = bias;

    c = __builtin_amdgcn_wmma_f32_16x16x32_f16(false, a, false, b, (short)0, c,
                                               false, false);

    // D: lane (half,m) VGPR r -> row tile*16 + half*8 + r, col m
    if (m < OUTP) {
      int row0 = tile * 16 + (half ? 8 : 0);
#pragma unroll
      for (int r = 0; r < 8; ++r) out[(size_t)(row0 + r) * OUTP + m] = c[r];
    }
  }
}

// ---------------- launcher ----------------

extern "C" void kernel_launch(void* const* d_in, const int* in_sizes, int n_in,
                              void* d_out, int out_size, void* d_ws, size_t ws_size,
                              hipStream_t stream) {
  (void)in_sizes; (void)n_in; (void)out_size; (void)ws_size;
  const float*     X    = (const float*)d_in[0];
  const long long* ei   = (const long long*)d_in[1];   // edge_index int64 [2,E]
  const float*     czW  = (const float*)d_in[3];
  const float*     czb  = (const float*)d_in[4];
  // d_in[5..6]: conv_r_* — dead (H0==0 => R unused)
  const float*     chW  = (const float*)d_in[7];
  const float*     chb  = (const float*)d_in[8];
  const float*     lzW  = (const float*)d_in[9];
  const float*     lzb  = (const float*)d_in[10];
  // d_in[11..12]: lin_r_* — dead
  const float*     lhW  = (const float*)d_in[13];
  const float*     lhb  = (const float*)d_in[14];
  const float*     att  = (const float*)d_in[15];
  const float*     outW = (const float*)d_in[16];
  const float*     outB = (const float*)d_in[17];
  float* out = (float*)d_out;

  float* ws  = (float*)d_ws;
  float* deg = ws;                       // [N]  (deg, then overwritten with dinv)
  float* AX  = ws + N_NODES;             // [N*24]
  float* C   = ws + 25 * N_NODES;        // [204] fused constants

  const int T = 256;
  const int gN = (N_NODES + T - 1) / T;
  const int gE = (N_EDGES + T - 1) / T;

  k_init_deg<<<gN, T, 0, stream>>>(deg);
  k_deg_edges<<<gE, T, 0, stream>>>(ei, deg);
  k_dinv_selfinit<<<gN, T, 0, stream>>>(X, deg, AX);
  k_scatter<<<gE, T, 0, stream>>>(ei, X, deg, AX);
  k_consts<<<1, 64, 0, stream>>>(czW, czb, chW, chb, lzW, lzb, lhW, lhb, att, C);

  const int tiles = N_NODES / 16;                       // 1250
  const int wavesPerBlock = T / 32;                     // 8
  const int gD = (tiles + wavesPerBlock - 1) / wavesPerBlock;
  k_dense<<<gD, T, 0, stream>>>(AX, C, outW, outB, out);
}
